// PAMCharEcoder_90503550861518
// MI455X (gfx1250) — matrix operand order repaired and checked by reading, not compiled, not run on hardware
//
#include <hip/hip_runtime.h>
#include <math.h>

typedef __attribute__((ext_vector_type(16))) _Float16 v16h;
typedef __attribute__((ext_vector_type(8)))  _Float16 v8h;
typedef __attribute__((ext_vector_type(8)))  float    v8f;

constexpr int Bb = 8, Ss = 512, Dd = 72;
constexpr int KP = 96, NP = 80;   // zero-padded K,N for the 72x72 projection GEMMs

// ---------------- positional encoding: pe[s,2k]=sin(s*div_k), pe[s,2k+1]=cos ----------------
__global__ void k_pe(float* pe) {
    int s = blockIdx.x * blockDim.x + threadIdx.x;
    if (s >= Ss) return;
    #pragma unroll 4
    for (int k = 0; k < Dd / 2; ++k) {
        // div = exp(-2k*ln(10000)/72) = exp(-0.255842788*k)
        float div = __expf(-0.255842788f * (float)k);
        float ang = (float)s * div;
        pe[s * Dd + 2 * k]     = __sinf(ang);
        pe[s * Dd + 2 * k + 1] = __cosf(ang);
    }
}

// ---------------- W (72x72) -> zero-padded 96x80, pre-swizzled into WMMA B-fragment layout ----
// B layout as packed here (contiguous K, matching the A side): lanes 0-15 hold N=lane, K=0..15 of chunk; lanes 16-31 hold K=16..31.
__global__ void k_swizzleW(const float* __restrict__ W, _Float16* __restrict__ Wsw) {
    int id = blockIdx.x * blockDim.x + threadIdx.x;
    if (id >= KP * NP) return;
    int kp = id / NP, np = id % NP;
    int nt = np >> 4, nn = np & 15;
    int kc = kp >> 5, kl = kp & 31;
    int lane = nn + (kl & 16);
    int e = kl & 15;
    float v = (kp < Dd && np < Dd) ? W[kp * Dd + np] : 0.f;
    Wsw[(((size_t)(nt * 3 + kc) * 32 + lane) << 4) + e] = (_Float16)v;
}

// ---------------- embedding gather + build Y_h (masked values | mask cols), pad zeroing -------
__global__ void k_embed(const int* __restrict__ ids, const unsigned char* __restrict__ mask,
                        const float* __restrict__ emb, float* __restrict__ xf,
                        _Float16* __restrict__ Ysw, _Float16* __restrict__ attP,
                        _Float16* __restrict__ v1P) {
    int id = blockIdx.x * blockDim.x + threadIdx.x;
    if (id >= Bb * Ss) return;
    int b = id / Ss, s = id % Ss;
    int tok = ids[id];
    const float* erow = emb + (size_t)tok * Dd;
    bool m = mask[id] != 0;
    int kc = s >> 5, kl = s & 31;
    int laneg = kl & 16, e = kl & 15;
    for (int h = 0; h < Dd; ++h) {
        float xv = erow[h];
        xf[(size_t)id * Dd + h] = xv;
        size_t base = ((size_t)(h * 16 + kc) * 32) << 4;       // B-fragment slab for (h, K-chunk)
        Ysw[base + (size_t)(laneg + b) * 16 + e]     = (_Float16)(m ? xv : 0.f);  // numerator col b
        Ysw[base + (size_t)(laneg + 8 + b) * 16 + e] = (_Float16)(m ? 1.f : 0.f); // denominator col
    }
    for (int p = Dd; p < KP; ++p) {                            // rewrite pads every call
        attP[(size_t)id * KP + p] = (_Float16)0.f;
        v1P[(size_t)id * KP + p]  = (_Float16)0.f;
    }
}

// ---------------- attention: per (head, i-tile) wave computes [16 x 16] = E_h @ Y_h ----------
// All 8 waves of a block share one head h (8 consecutive tasks never straddle a 32-task head
// boundary), so the pe column is staged once per block. A-fragment elements exp(pe_i*pe_j) are
// computed on the fly as exp2((pe_i*log2e)*pe_j) with the RAW v_exp_f32 builtin: arguments are
// bounded by |log2e|, so no denormal guard is needed -> exactly one v_mul + one v_exp_f32 per
// element. The next chunk's B-fragment load is issued before the exp burst so TRANS work hides
// the load latency ahead of each v_wmma.
__global__ void __launch_bounds__(256) k_attn(const float* __restrict__ pe,
                                              const _Float16* __restrict__ Ysw,
                                              _Float16* __restrict__ attP) {
    __shared__ float lpe[Ss];                       // one pe column, shared by the whole block
    int wid = threadIdx.x >> 5, lane = threadIdx.x & 31;
    int h = blockIdx.x >> 2;                        // 4 blocks (32 i-tiles) per head
    int i0 = (((blockIdx.x & 3) << 3) + wid) << 4;  // this wave's i-tile
    const _Float16* Yh = Ysw + (((size_t)h * 16) * 32 << 4);
    __builtin_prefetch((const void*)Yh, 0, 1);      // global_prefetch of this head's Y slab
    for (int idx = threadIdx.x; idx < Ss; idx += 256) lpe[idx] = pe[idx * Dd + h];
    __syncthreads();

    int mrow = lane & 15, hi16 = lane >> 4;
    float pei2 = lpe[i0 + mrow] * 1.44269504088896f;  // pe_i * log2(e), loop-invariant
    v8f acc = {};
    v16h bf = *(const v16h*)(Yh + ((size_t)lane << 4));           // chunk 0 B-fragment
    for (int kc = 0; kc < 16; ++kc) {
        v16h bfn = bf;
        if (kc < 15)                                              // issue next chunk's load early
            bfn = *(const v16h*)(Yh + (((size_t)(kc + 1) * 32 + lane) << 4));
        int j0 = kc << 5;
        v16h a;
        #pragma unroll
        for (int e = 0; e < 16; ++e) {
            int k = e + (hi16 << 4);                // 16-bit A-matrix K index for this element
            a[e] = (_Float16)__builtin_amdgcn_exp2f(pei2 * lpe[j0 + k]);  // v_mul + v_exp_f32
        }
        acc = __builtin_amdgcn_wmma_f32_16x16x32_f16(false, a, false, bf, (short)0, acc,
                                                     false, false);
        bf = bfn;
    }
    int ncol = lane & 15;                           // C layout: N=lane%16, M=r+8*(lane/16)
    #pragma unroll
    for (int r = 0; r < 8; ++r) {
        float num = acc[r];
        float den = __shfl(num, (hi16 << 4) + ((ncol + 8) & 15), 32);
        float ratio = num * __builtin_amdgcn_rcpf(den);           // v_rcp_f32, no IEEE-div expansion
        if (ncol < 8) {                             // cols 0..7 = numerators, 8..15 = denominators
            int i = i0 + r + (hi16 << 3);
            attP[((size_t)(ncol * Ss + i)) * KP + h] = (_Float16)ratio;
        }
    }
}

// ---------------- fused GEMM + bias + relu: [4096 x 96] @ pre-swizzled [96 x 80] -------------
__global__ void __launch_bounds__(256) k_gemm(const _Float16* __restrict__ Ain,
                                              const _Float16* __restrict__ Wsw,
                                              const float* __restrict__ bias,
                                              _Float16* __restrict__ outH,
                                              float* __restrict__ outF, int mode) {
    int wid = threadIdx.x >> 5, lane = threadIdx.x & 31;
    int task = blockIdx.x * 8 + wid;                // 256 M-tiles * 5 N-tiles = 1280 tasks
    int nt = task / 256;
    int m0 = (task % 256) << 4;
    int hi16 = lane >> 4;
    int mrow = m0 + (lane & 15);
    v8f acc = {};
    #pragma unroll
    for (int kc = 0; kc < 3; ++kc) {
        int kbase = (kc << 5) + (hi16 << 4);        // A layout: K=kbase+e (contiguous)
        v8h lo = *(const v8h*)(Ain + (size_t)mrow * KP + kbase);
        v8h hv = *(const v8h*)(Ain + (size_t)mrow * KP + kbase + 8);
        v16h a = __builtin_shufflevector(lo, hv, 0,1,2,3,4,5,6,7,8,9,10,11,12,13,14,15);
        v16h bf = *(const v16h*)(Wsw + (((size_t)(nt * 3 + kc) * 32 + lane) << 4));
        acc = __builtin_amdgcn_wmma_f32_16x16x32_f16(false, a, false, bf, (short)0, acc,
                                                     false, false);
    }
    int n = (nt << 4) + (lane & 15);
    if (n < Dd) {
        float bn = bias[n];
        #pragma unroll
        for (int r = 0; r < 8; ++r) {
            int row = m0 + r + (hi16 << 3);
            float v = fmaxf(acc[r] + bn, 0.f);
            if (mode == 0) outH[(size_t)row * KP + n] = (_Float16)v;  // feeds GEMM2 (pads pre-zeroed)
            else           outF[(size_t)row * Dd + n] = v;            // f32 for LayerNorm
        }
    }
}

// ---------------- LayerNorm(ddof=1) + residual + masked pool + L2 normalize (deterministic) ---
__global__ void __launch_bounds__(256) k_final(const float* __restrict__ cbuf,
                                               const float* __restrict__ xf,
                                               const unsigned char* __restrict__ mask,
                                               const float* __restrict__ gamma,
                                               const float* __restrict__ beta,
                                               float* __restrict__ outp) {
    __shared__ float smean[Ss], srstd[Ss], red[256], pool[Dd], snorm;
    int b = blockIdx.x, t = threadIdx.x;
    for (int s = t; s < Ss; s += 256) {
        const float* crow = cbuf + ((size_t)(b * Ss + s)) * Dd;
        float sum = 0.f, sq = 0.f;
        for (int d = 0; d < Dd; ++d) { float c = crow[d]; sum += c; sq += c * c; }
        float mean = sum / (float)Dd;
        float var = (sq - (float)Dd * mean * mean) / (float)(Dd - 1);
        smean[s] = mean;
        srstd[s] = 1.f / (sqrtf(fmaxf(var, 0.f)) + 1e-6f);
    }
    __syncthreads();
    for (int d = 0; d < Dd; ++d) {
        float g = gamma[d], be = beta[d], part = 0.f;
        for (int s = t; s < Ss; s += 256) {
            if (mask[b * Ss + s]) {
                size_t off = ((size_t)(b * Ss + s)) * Dd + d;
                float ln = g * (cbuf[off] - smean[s]) * srstd[s] + be;
                part += xf[off] + ln;
            }
        }
        red[t] = part;
        __syncthreads();
        for (int off = 128; off > 0; off >>= 1) {   // fixed-order tree: bit-deterministic
            if (t < off) red[t] += red[t + off];
            __syncthreads();
        }
        if (t == 0) pool[d] = red[0];
        __syncthreads();
    }
    if (t == 0) {
        float nrm = 0.f;
        for (int d = 0; d < Dd; ++d) nrm += pool[d] * pool[d];
        snorm = sqrtf(nrm);
    }
    __syncthreads();
    if (t < Dd) outp[b * Dd + t] = pool[t] / snorm;
}

extern "C" void kernel_launch(void* const* d_in, const int* in_sizes, int n_in,
                              void* d_out, int out_size, void* d_ws, size_t ws_size,
                              hipStream_t stream) {
    (void)in_sizes; (void)n_in; (void)out_size; (void)ws_size;
    const int*           ids  = (const int*)d_in[0];
    const unsigned char* mask = (const unsigned char*)d_in[1];
    const float* emb  = (const float*)d_in[2];
    const float* W1   = (const float*)d_in[3];
    const float* b1   = (const float*)d_in[4];
    const float* W2   = (const float*)d_in[5];
    const float* b2   = (const float*)d_in[6];
    const float* gam  = (const float*)d_in[7];
    const float* bet  = (const float*)d_in[8];

    char* ws = (char*)d_ws;                         // ~5.3 MB total, all 256B-aligned
    float*    pe    = (float*)(ws + 0);             // 512*72*4      = 147456
    float*    xf    = (float*)(ws + 147456);        // 8*512*72*4    = 1179648
    _Float16* Ysw   = (_Float16*)(ws + 1327104);    // 72*16*32*16*2 = 1179648
    _Float16* attP  = (_Float16*)(ws + 2506752);    // 4096*96*2     = 786432
    _Float16* v1P   = (_Float16*)(ws + 3293184);    // 4096*96*2     = 786432
    float*    cbuf  = (float*)(ws + 4079616);       // 4096*72*4     = 1179648
    _Float16* W1sw  = (_Float16*)(ws + 5259264);    // 5*3*32*16*2   = 15360
    _Float16* W2sw  = (_Float16*)(ws + 5274624);    // 15360
    float*    outp  = (float*)d_out;

    k_pe      <<<2,   256, 0, stream>>>(pe);
    k_swizzleW<<<30,  256, 0, stream>>>(W1, W1sw);
    k_swizzleW<<<30,  256, 0, stream>>>(W2, W2sw);
    k_embed   <<<16,  256, 0, stream>>>(ids, mask, emb, xf, Ysw, attP, v1P);
    k_attn    <<<288, 256, 0, stream>>>(pe, Ysw, attP);            // 2304 wave-tiles, 16 WMMA each
    k_gemm    <<<160, 256, 0, stream>>>(attP, W1sw, b1, v1P, nullptr, 0);
    k_gemm    <<<160, 256, 0, stream>>>(v1P,  W2sw, b2, nullptr, cbuf, 1);
    k_final   <<<8,   256, 0, stream>>>(cbuf, xf, mask, gam, bet, outp);
}
